// GAT_22462678958399
// MI455X (gfx1250) — compile-verified, matrix operand order checked by reading, not run on hardware
//
#include <hip/hip_runtime.h>
#include <hip/hip_bf16.h>
#include <math.h>

// ---------------- problem constants ----------------
#define NN      116          // nodes per graph
#define NSQ     13456        // 116*116
#define KP      13472        // NSQ padded to multiple of 32 (= 421*32)
#define BATCH   4096
#define HOUT    1024
#define ALPHA   0.2f
#define NEG_INF (-9e15f)

// ---------------- WMMA vector types ----------------
typedef __attribute__((ext_vector_type(16))) __bf16 v16bf;
typedef __attribute__((ext_vector_type(8)))  __bf16 v8bf;
typedef __attribute__((ext_vector_type(8)))  float  v8f;

// explicit global (addrspace 1) access so the backend emits
// global_load_b128 / global_store_b32 (LOADcnt/STOREcnt only)
// instead of flat_* (which also ties up DScnt / the LDS port).
typedef const __attribute__((address_space(1))) v8bf*  gc_v8bf;
typedef       __attribute__((address_space(1))) float* g_f32;

static __device__ inline v8bf gload8(const __bf16* p) {
    return *(gc_v8bf)(unsigned long long)p;
}
static __device__ inline void gstoref(float* p, float v) {
    *(g_f32)(unsigned long long)p = v;
}

static __device__ inline v16bf combine16(v8bf lo, v8bf hi) {
    v16bf r;
#pragma unroll
    for (int i = 0; i < 8; ++i) { r[i] = lo[i]; r[i + 8] = hi[i]; }
    return r;
}

// ====================================================================
// Kernel 1: u1 = W_att @ a1, u2 = W_att @ a2   (116-element vectors)
// ====================================================================
__global__ void u_kernel(const float* __restrict__ Watt,
                         const float* __restrict__ a1,
                         const float* __restrict__ a2,
                         float* __restrict__ u1, float* __restrict__ u2) {
    int t = threadIdx.x;
    if (t < NN) {
        float s1 = 0.f, s2 = 0.f;
#pragma unroll 8
        for (int h = 0; h < 64; ++h) {
            float w = Watt[t * 64 + h];
            s1 += w * a1[h];
            s2 += w * a2[h];
        }
        u1[t] = s1;
        u2[t] = s2;
    }
}

// ====================================================================
// Kernel 2: per-graph attention + masked softmax -> bf16 rows (KP stride)
// one block (128 threads / 4 waves) per batch element; all global
// traffic coalesced (lane = feature / column index).
// ====================================================================
__global__ void __launch_bounds__(128)
att_kernel(const float* __restrict__ x, const float* __restrict__ adj,
           const float* __restrict__ u1, const float* __restrict__ u2,
           __bf16* __restrict__ att) {
    const int b    = blockIdx.x;
    const int tid  = threadIdx.x;
    const int lane = tid & 31;
    const int wave = tid >> 5;

    __shared__ float su1[128], su2[128], sf1[128], sf2[128];
    if (tid < NN) { su1[tid] = u1[tid]; su2[tid] = u2[tid]; }
    __syncthreads();

    const float* xb = x + (size_t)b * NSQ;

    // f1[i] = x[b,i,:] . u1 ; f2[i] = x[b,i,:] . u2   (wave per row)
    for (int i = wave; i < NN; i += 4) {
        const float* row = xb + i * NN;
        float s1 = 0.f, s2 = 0.f;
#pragma unroll
        for (int c = 0; c < 4; ++c) {
            int f = c * 32 + lane;
            if (f < NN) {
                float v = row[f];
                s1 += v * su1[f];
                s2 += v * su2[f];
            }
        }
#pragma unroll
        for (int off = 16; off; off >>= 1) {
            s1 += __shfl_xor(s1, off);
            s2 += __shfl_xor(s2, off);
        }
        if (lane == 0) { sf1[i] = s1; sf2[i] = s2; }
    }
    __syncthreads();

    const float*  adjb = adj + (size_t)b * NSQ;
    __bf16*       attb = att + (size_t)b * KP;

    // masked leaky-relu scores + row softmax (wave per row)
    for (int i = wave; i < NN; i += 4) {
        const float f1i = sf1[i];
        float ev[4];
        float m = -3.0e38f;
#pragma unroll
        for (int c = 0; c < 4; ++c) {
            int j = c * 32 + lane;
            float e = -3.0e38f;
            if (j < NN) {
                float t = f1i + sf2[j];
                t = (t > 0.f) ? t : ALPHA * t;
                e = (adjb[i * NN + j] > 0.f) ? t : NEG_INF;
            }
            ev[c] = e;
            m = fmaxf(m, e);
        }
#pragma unroll
        for (int off = 16; off; off >>= 1) m = fmaxf(m, __shfl_xor(m, off));
        float s = 0.f;
        float p[4];
#pragma unroll
        for (int c = 0; c < 4; ++c) {
            int j = c * 32 + lane;
            float pv = (j < NN) ? expf(ev[c] - m) : 0.f;
            p[c] = pv;
            s += pv;
        }
#pragma unroll
        for (int off = 16; off; off >>= 1) s += __shfl_xor(s, off);
        float inv = 1.f / s;
#pragma unroll
        for (int c = 0; c < 4; ++c) {
            int j = c * 32 + lane;
            if (j < NN) attb[i * NN + j] = (__bf16)(p[c] * inv);
        }
    }
    // zero the K padding (NSQ..KP) so it contributes nothing to the GEMM
    if (tid < (KP - NSQ)) attb[NSQ + tid] = (__bf16)0.f;
}

// ====================================================================
// Kernel 3: W1 [13456,1024] f32  ->  W1^T [1024, KP] bf16 (LDS transpose)
// grid (421, 32), block (32, 8); k >= NSQ rows become zeros (padding).
// ====================================================================
__global__ void w1t_kernel(const float* __restrict__ W1, __bf16* __restrict__ w1t) {
    __shared__ float tile[32][33];
    const int tk0 = blockIdx.x * 32;   // k tile
    const int tn0 = blockIdx.y * 32;   // n tile
    const int tx = threadIdx.x, ty = threadIdx.y;
#pragma unroll
    for (int r = 0; r < 4; ++r) {
        int k = tk0 + ty + 8 * r;
        int n = tn0 + tx;
        tile[ty + 8 * r][tx] = (k < NSQ) ? W1[(size_t)k * HOUT + n] : 0.f;
    }
    __syncthreads();
#pragma unroll
    for (int r = 0; r < 4; ++r) {
        int n = tn0 + ty + 8 * r;
        int k = tk0 + tx;
        w1t[(size_t)n * KP + k] = (__bf16)tile[tx][ty + 8 * r];
    }
}

// ====================================================================
// Kernel 4: h1 = relu(att[4096,KP](bf16) @ W1^T[1024,KP](bf16)^T + b1)
// WMMA bf16 16x16x32, f32 accumulate.
//
// Block = 128(M) x 256(N) tile, 256 threads = 8 waves (2 along M x 4
// along N); each wave owns a 64x64 region = 4x4 WMMA tiles:
//   16 v_wmma per K-step against 16 global_load_b128 (ratio 1:1).
// Single base pointer per operand stream; the per-fragment displacement
// (mi*16*KP*2 = 431104*mi bytes < 2^23) folds into the 24-bit IOFFSET.
//
// Fragment addressing follows the CDNA5 16-bit A/B layout:
//   lane&15 -> M (A) / N (B) index; lane>>4 selects K half;
//   element e -> k = e + (e>=8 ? 8 : 0) + (lane>>4)*8  (two 16B chunks).
// ====================================================================
__global__ void __launch_bounds__(256)
gemm_kernel(const __bf16* __restrict__ A,    // [BATCH, KP]
            const __bf16* __restrict__ Bt,   // [HOUT, KP]
            const float*  __restrict__ bias, // [HOUT]
            float* __restrict__ h1) {        // [BATCH, HOUT]
    const int tid   = threadIdx.x;
    const int lane  = tid & 31;
    const int wave  = tid >> 5;
    const int lm    = lane & 15;
    const int khalf = lane >> 4;

    const int m0 = blockIdx.y * 128 + (wave & 1) * 64;   // 4 x 16 rows
    const int n0 = blockIdx.x * 256 + (wave >> 1) * 64;  // 4 x 16 cols

    v8f c[4][4];
#pragma unroll
    for (int mi = 0; mi < 4; ++mi)
#pragma unroll
        for (int ni = 0; ni < 4; ++ni) c[mi][ni] = (v8f)(0.f);

    // one live pointer per stream; fragment row displacement is constant
    const __bf16* aBase = A  + (size_t)(m0 + lm) * KP + khalf * 8;
    const __bf16* bBase = Bt + (size_t)(n0 + lm) * KP + khalf * 8;

    for (int k = 0; k < KP; k += 32) {
        v16bf af[4], bf[4];
#pragma unroll
        for (int mi = 0; mi < 4; ++mi) {
            v8bf lo = gload8(aBase + mi * (16 * KP));
            v8bf hi = gload8(aBase + mi * (16 * KP) + 16);
            af[mi] = combine16(lo, hi);
        }
#pragma unroll
        for (int ni = 0; ni < 4; ++ni) {
            v8bf lo = gload8(bBase + ni * (16 * KP));
            v8bf hi = gload8(bBase + ni * (16 * KP) + 16);
            bf[ni] = combine16(lo, hi);
        }
        aBase += 32;
        bBase += 32;
#pragma unroll
        for (int mi = 0; mi < 4; ++mi)
#pragma unroll
            for (int ni = 0; ni < 4; ++ni)
                c[mi][ni] = __builtin_amdgcn_wmma_f32_16x16x32_bf16(
                    false, af[mi], false, bf[ni], (short)0, c[mi][ni],
                    false, false);
    }

    // epilogue: bias + relu.  C layout: M = khalf*8 + r, N = lane&15.
#pragma unroll
    for (int ni = 0; ni < 4; ++ni) {
        const int n = n0 + 16 * ni + lm;
        const float bv = bias[n];
#pragma unroll
        for (int mi = 0; mi < 4; ++mi) {
            v8f cc = c[mi][ni];
#pragma unroll
            for (int r = 0; r < 8; ++r) {
                int m = m0 + 16 * mi + khalf * 8 + r;
                float v = cc[r] + bv;
                gstoref(&h1[(size_t)m * HOUT + n], (v > 0.f) ? v : 0.f);
            }
        }
    }
}

// ====================================================================
// Kernel 5: h2 = relu(h1 @ W2 + b2); out = log_softmax(h2) over 2 classes
// one wave per batch row.
// ====================================================================
__global__ void __launch_bounds__(128)
head_kernel(const float* __restrict__ h1, const float* __restrict__ W2,
            const float* __restrict__ b2, float* __restrict__ out) {
    const int tid  = threadIdx.x;
    const int lane = tid & 31;
    const int wave = tid >> 5;
    const int b    = blockIdx.x * 4 + wave;

    const float* row = h1 + (size_t)b * HOUT;
    float a0 = 0.f, a1 = 0.f;
#pragma unroll 4
    for (int c = 0; c < 32; ++c) {
        int k = c * 32 + lane;
        float v = row[k];
        a0 += v * W2[k * 2 + 0];
        a1 += v * W2[k * 2 + 1];
    }
#pragma unroll
    for (int off = 16; off; off >>= 1) {
        a0 += __shfl_xor(a0, off);
        a1 += __shfl_xor(a1, off);
    }
    if (lane == 0) {
        float v0 = a0 + b2[0]; v0 = (v0 > 0.f) ? v0 : 0.f;
        float v1 = a1 + b2[1]; v1 = (v1 > 0.f) ? v1 : 0.f;
        float m  = fmaxf(v0, v1);
        float lse = m + logf(expf(v0 - m) + expf(v1 - m));
        out[b * 2 + 0] = v0 - lse;
        out[b * 2 + 1] = v1 - lse;
    }
}

// ====================================================================
// launch
// ====================================================================
extern "C" void kernel_launch(void* const* d_in, const int* in_sizes, int n_in,
                              void* d_out, int out_size, void* d_ws, size_t ws_size,
                              hipStream_t stream) {
    const float* x    = (const float*)d_in[0];   // [B, N, N]
    const float* adj  = (const float*)d_in[1];   // [B, N, N]
    const float* Watt = (const float*)d_in[2];   // [N, 64]
    const float* a1   = (const float*)d_in[3];   // [64]
    const float* a2   = (const float*)d_in[4];   // [64]
    const float* W1   = (const float*)d_in[5];   // [N*N, 1024]
    const float* b1   = (const float*)d_in[6];   // [1024]
    const float* W2   = (const float*)d_in[7];   // [1024, 2]
    const float* b2   = (const float*)d_in[8];   // [2]
    float* out        = (float*)d_out;           // [B, 2]

    // workspace layout (256B aligned offsets), total ~155 MB
    char* ws = (char*)d_ws;
    float*  u1  = (float*)(ws + 0);                    //   512 B
    float*  u2  = (float*)(ws + 512);                  //   512 B
    __bf16* w1t = (__bf16*)(ws + 1024);                // 1024*KP*2  = 27,590,656 B
    __bf16* att = (__bf16*)(ws + 1024 + (size_t)HOUT * KP * 2);          // B*KP*2 = 110,362,624 B
    float*  h1  = (float*)(ws + 1024 + (size_t)HOUT * KP * 2
                                     + (size_t)BATCH * KP * 2);          // B*1024*4 = 16,777,216 B

    // 1) fold W_att into two 116-vectors
    u_kernel<<<1, 128, 0, stream>>>(Watt, a1, a2, u1, u2);

    // 2) attention + masked softmax -> bf16 (one block per graph)
    att_kernel<<<BATCH, 128, 0, stream>>>(x, adj, u1, u2, att);

    // 3) W1 -> bf16 transposed [1024, KP]
    w1t_kernel<<<dim3(KP / 32, HOUT / 32), dim3(32, 8), 0, stream>>>(W1, w1t);

    // 4) big WMMA GEMM + bias + relu  (128x256 block tile, 64x64 per wave)
    gemm_kernel<<<dim3(HOUT / 256, BATCH / 128), 256, 0, stream>>>(att, w1t, b1, h1);

    // 5) head + log_softmax
    head_kernel<<<BATCH / 4, 128, 0, stream>>>(h1, W2, b2, out);
}
